// bSpNCN_9285719294381
// MI455X (gfx1250) — matrix operand order repaired and checked by reading, not compiled, not run on hardware
//
#include <hip/hip_runtime.h>

// ---------------------------------------------------------------------------
// bSpNCN step for MI455X (gfx1250, wave32, WMMA).
// All GEMMs are C = A[M,K] @ W[N,K]^T with A,W row-major (K contiguous) ->
// mapped onto v_wmma_f32_16x16x32_bf16 with f32 accumulation.
// Double-buffered LDS, register-staged global loads, one barrier per K-step.
// ---------------------------------------------------------------------------

typedef unsigned short u16;
typedef unsigned int   u32;
typedef __attribute__((ext_vector_type(16))) __bf16 v16bf;
typedef __attribute__((ext_vector_type(8)))  float  v8f;
typedef __attribute__((ext_vector_type(4)))  u32    u32x4;

__device__ __forceinline__ u16 f32_to_bf16_rne(float f) {
    u32 u = __float_as_uint(f);
    u32 r = u + 0x7FFFu + ((u >> 16) & 1u);   // round-to-nearest-even
    return (u16)(r >> 16);
}

union FragU { u32x4 q[2]; v16bf v; };

// --------------------------- f32 -> bf16 convert ---------------------------
// Destination has arbitrary row stride so weight matrices can be packed
// K-concatenated (WC1 = [Eg0 | Ed1], WC2 = [Eg1 | Ed2]). cols >= 256 always.
__global__ void cvt_f32_bf16(const float* __restrict__ src, u16* __restrict__ dst,
                             int rows, int cols, int ldDst) {
    for (int r = blockIdx.x; r < rows; r += gridDim.x) {
        const float* s = src + (size_t)r * cols;
        u16* d = dst + (size_t)r * ldDst;
        for (int c = threadIdx.x; c < cols; c += blockDim.x)
            d[c] = f32_to_bf16_rne(s[c]);
    }
}

// ------------------------------ WMMA GEMM ----------------------------------
// Block tile 128x128, 256 threads = 8 waves (4 M-waves x 2 N-waves), each
// wave 32x64 = 2x4 fragments of v_wmma_f32_16x16x32_bf16. BK=32, LDS row
// stride padded to 40 halfs. Requires: M%128==0, N%128==0, K%32==0.
//
// Epilogue: res = sAcc*acc + sBase*base[gr,gc]
//   outBf  : bf16(res) at row stride ldBf            (K-concat error buffers)
//   outF32 : res + base2[gr,gc] at row stride ldF32  (in-place running sums)
//   outLif : full LIF update, scatters j/v/x/s into [B,4096] output layout
__global__ __launch_bounds__(256)
void wmma_gemm_bf16(const u16* __restrict__ A, const u16* __restrict__ W,
                    int K, int N,
                    const float* __restrict__ base, int ldBase,
                    float sBase, float sAcc,
                    u16* __restrict__ outBf, int ldBf,
                    float* __restrict__ outF32, const float* __restrict__ base2, int ldF32,
                    const float* __restrict__ jin, const float* __restrict__ vin,
                    const float* __restrict__ xin,
                    float* __restrict__ outLif, int lifOff) {
    __shared__ u16 sA[2][128 * 40];
    __shared__ u16 sW[2][128 * 40];

    const int tid   = threadIdx.x;
    const int lane  = tid & 31;
    const int wave  = tid >> 5;
    const int waveM = wave & 3;    // 0..3
    const int waveN = wave >> 2;   // 0..1
    const int rowBase = blockIdx.y * 128;
    const int colBase = blockIdx.x * 128;

    // cooperative tile loaders: thread t stages 16 halfs of A-tile and W-tile
    const int lr = tid >> 1;          // 0..127
    const int lc = (tid & 1) * 16;    // 0 or 16
    const u16* gA = A + (size_t)(rowBase + lr) * K + lc;
    const u16* gW = W + (size_t)(colBase + lr) * K + lc;
    const int stOff = lr * 40 + lc;

    const int r15 = lane & 15;
    const int hi  = lane >> 4;
    // A fragment (16x32 bf16): lanes 0-15 hold K 0-7 / 16-23, lanes 16-31 K 8-15 / 24-31
    const int ldsOffA = (waveM * 32 + r15) * 40 + hi * 8;
    // B fragment (32x16 bf16): lanes 0-15 hold K 0-7 / 16-23, lanes 16-31 K 8-15 / 24-31
    const int ldsOffW = (waveN * 64 + r15) * 40 + hi * 8;

    v8f acc[2][4] = {};

    // ---- prologue: stage K-step 0 into LDS buffer 0 ----
    u32x4 ra0 = *(const u32x4*)(gA);
    u32x4 ra1 = *(const u32x4*)(gA + 8);
    u32x4 rw0 = *(const u32x4*)(gW);
    u32x4 rw1 = *(const u32x4*)(gW + 8);
    {
        u16* pa = &sA[0][stOff];
        u16* pw = &sW[0][stOff];
        *(u32x4*)(pa) = ra0; *(u32x4*)(pa + 8) = ra1;
        *(u32x4*)(pw) = rw0; *(u32x4*)(pw + 8) = rw1;
    }
    __syncthreads();

    int cur = 0;
    for (int k0 = 0; k0 < K; k0 += 32) {
        const bool more = (k0 + 32) < K;
        if (more) {
            // stage next K-step into registers (overlaps with WMMA below)
            const u32x4* na = (const u32x4*)(gA + k0 + 32);
            const u32x4* nw = (const u32x4*)(gW + k0 + 32);
            ra0 = na[0]; ra1 = na[1];
            rw0 = nw[0]; rw1 = nw[1];
            if (k0 + 64 < K) {   // warm L2/WGP$ two steps ahead
                __builtin_prefetch(gA + k0 + 64, 0, 3);
                __builtin_prefetch(gW + k0 + 64, 0, 3);
            }
        }

        const u16* aB = &sA[cur][ldsOffA];
        const u16* wB = &sW[cur][ldsOffW];
        FragU aF[2], wF[4];
#pragma unroll
        for (int mi = 0; mi < 2; ++mi) {
            const u16* p = aB + mi * 16 * 40;
            aF[mi].q[0] = *(const u32x4*)(p);
            aF[mi].q[1] = *(const u32x4*)(p + 16);
        }
#pragma unroll
        for (int ni = 0; ni < 4; ++ni) {
            const u16* p = wB + ni * 16 * 40;
            wF[ni].q[0] = *(const u32x4*)(p);
            wF[ni].q[1] = *(const u32x4*)(p + 16);
        }
#pragma unroll
        for (int mi = 0; mi < 2; ++mi)
#pragma unroll
            for (int ni = 0; ni < 4; ++ni)
                acc[mi][ni] = __builtin_amdgcn_wmma_f32_16x16x32_bf16(
                    false, aF[mi].v, false, wF[ni].v,
                    (short)0, acc[mi][ni], false, false);

        if (more) {
            // store staged regs into the buffer nobody reads this step
            u16* pa = &sA[cur ^ 1][stOff];
            u16* pw = &sW[cur ^ 1][stOff];
            *(u32x4*)(pa) = ra0; *(u32x4*)(pa + 8) = ra1;
            *(u32x4*)(pw) = rw0; *(u32x4*)(pw + 8) = rw1;
        }
        __syncthreads();   // single barrier per K-step
        cur ^= 1;
    }

    // ------------------------------ epilogue -------------------------------
#pragma unroll
    for (int mi = 0; mi < 2; ++mi) {
#pragma unroll
        for (int ni = 0; ni < 4; ++ni) {
            const int gc = colBase + waveN * 64 + ni * 16 + r15;
#pragma unroll
            for (int e = 0; e < 8; ++e) {
                const int gr = rowBase + waveM * 32 + mi * 16 + hi * 8 + e;
                float res = sAcc * acc[mi][ni][e];
                if (base)  res += sBase * base[(size_t)gr * ldBase + gc];
                if (outBf) outBf[(size_t)gr * ldBf + gc] = f32_to_bf16_rne(res);
                if (outF32) {
                    float r2 = res;
                    if (base2) r2 += base2[(size_t)gr * ldF32 + gc];
                    outF32[(size_t)gr * ldF32 + gc] = r2;
                }
                if (outLif) {
                    const size_t si = (size_t)gr * 512 + gc;
                    float j = jin[si], v = vin[si], x = xin[si];
                    j = j + 0.1f * (-0.25f * j + res);   // DT/TAU_J=0.1, KAPPA_J=0.25
                    v = v + 0.05f * (-v + j);            // DT/TAU_M=0.05
                    const float s = (v > 0.4f) ? 1.0f : 0.0f;
                    v *= (1.0f - s);
                    x = x - x * 0.05f + s;               // X_DECAY=20
                    float* o = outLif + (size_t)gr * 4096 + lifOff + gc;
                    o[0]    = j;
                    o[512]  = v;
                    o[1024] = x;
                    o[1536] = s;
                }
            }
        }
    }
}

// ---------------------------------------------------------------------------
extern "C" void kernel_launch(void* const* d_in, const int* in_sizes, int n_in,
                              void* d_out, int out_size, void* d_ws, size_t ws_size,
                              hipStream_t stream) {
    (void)in_sizes; (void)n_in; (void)out_size; (void)ws_size;
    const int B = 8192;

    const float* x_input  = (const float*)d_in[0];
    const float* y_target = (const float*)d_in[1];
    const float* j1 = (const float*)d_in[2];
    const float* v1 = (const float*)d_in[3];
    const float* s1 = (const float*)d_in[4];
    const float* x1 = (const float*)d_in[5];
    const float* j2 = (const float*)d_in[6];
    const float* v2 = (const float*)d_in[7];
    const float* s2 = (const float*)d_in[8];
    const float* x2 = (const float*)d_in[9];
    const float* W0 = (const float*)d_in[10];
    const float* W1 = (const float*)d_in[11];
    const float* W2 = (const float*)d_in[12];
    const float* V0 = (const float*)d_in[13];
    const float* V1 = (const float*)d_in[14];
    const float* V2 = (const float*)d_in[15];
    const float* Eg0 = (const float*)d_in[16];
    const float* Eg1 = (const float*)d_in[17];
    // d_in[18] = Eg2, d_in[19] = Ed0 : unused by the reference
    const float* Ed1 = (const float*)d_in[20];
    const float* Ed2 = (const float*)d_in[21];
    float* out = (float*)d_out;

    // ----- workspace carve-out (256B aligned) -----
    char* ws = (char*)d_ws;
    size_t off = 0;
    auto alloc = [&](size_t bytes) -> char* {
        char* p = ws + off;
        off += (bytes + 255) & ~(size_t)255;
        return p;
    };
    u16* s1b  = (u16*)alloc((size_t)B * 512 * 2);
    u16* s2b  = (u16*)alloc((size_t)B * 512 * 2);
    u16* xinb = (u16*)alloc((size_t)B * 1024 * 2);
    u16* ytb  = (u16*)alloc((size_t)B * 256 * 2);
    u16* W0b  = (u16*)alloc((size_t)1024 * 512 * 2);
    u16* W1b  = (u16*)alloc((size_t)512 * 512 * 2);
    u16* W2b  = (u16*)alloc((size_t)512 * 256 * 2);
    u16* V0b  = (u16*)alloc((size_t)512 * 1024 * 2);
    u16* V1b  = (u16*)alloc((size_t)512 * 512 * 2);
    u16* V2b  = (u16*)alloc((size_t)256 * 512 * 2);
    u16* WC1  = (u16*)alloc((size_t)512 * 1536 * 2);  // [Eg0 | Ed1]
    u16* WC2  = (u16*)alloc((size_t)512 * 768 * 2);   // [Eg1 | Ed2]
    u16* EA1  = (u16*)alloc((size_t)B * 1536 * 2);    // [e_gen0 | e_disc2]
    u16* EA2  = (u16*)alloc((size_t)B * 768 * 2);     // [e_gen1 | e_disc3]
    float* S1 = (float*)alloc((size_t)B * 512 * 4);   // e_gen1 + e_disc1
    float* S2 = (float*)alloc((size_t)B * 512 * 4);   // e_gen2 + e_disc2

    // ----- f32 -> bf16 conversions -----
    auto cvt = [&](const float* s, u16* d, int r, int c, int ld) {
        int blocks = r < 4096 ? r : 4096;
        cvt_f32_bf16<<<blocks, dim3(256), 0, stream>>>(s, d, r, c, ld);
    };
    cvt(s1, s1b, B, 512, 512);
    cvt(s2, s2b, B, 512, 512);
    cvt(x_input, xinb, B, 1024, 1024);
    cvt(y_target, ytb, B, 256, 256);
    cvt(W0, W0b, 1024, 512, 512);
    cvt(W1, W1b, 512, 512, 512);
    cvt(W2, W2b, 512, 256, 256);
    cvt(V0, V0b, 512, 1024, 1024);
    cvt(V1, V1b, 512, 512, 512);
    cvt(V2, V2b, 256, 512, 512);
    cvt(Eg0, WC1, 512, 1024, 1536);
    cvt(Ed1, WC1 + 1024, 512, 512, 1536);
    cvt(Eg1, WC2, 512, 512, 768);
    cvt(Ed2, WC2 + 512, 512, 256, 768);

    // ----- GEMMs -----
    auto gemm = [&](const u16* Aa, const u16* Ww, int K, int N,
                    const float* baseP, int ldBase, float sB, float sAc,
                    u16* oBf, int ldBf,
                    float* oF, const float* b2, int ldF,
                    const float* jP, const float* vP, const float* xP,
                    float* oL, int lOff) {
        dim3 grid(N / 128, B / 128);
        wmma_gemm_bf16<<<grid, dim3(256), 0, stream>>>(
            Aa, Ww, K, N, baseP, ldBase, sB, sAc,
            oBf, ldBf, oF, b2, ldF, jP, vP, xP, oL, lOff);
    };

    // Phase 1: error neurons (res = base - A@W^T)
    // e_gen0  = x_input - s1@W0^T         -> bf16 into EA1[:,0:1024]
    gemm(s1b, W0b, 512, 1024, x_input, 1024, 1.f, -1.f,
         EA1, 1536, nullptr, nullptr, 0, nullptr, nullptr, nullptr, nullptr, 0);
    // e_disc1 = x1 - x_input@V0^T         -> f32 S1
    gemm(xinb, V0b, 1024, 512, x1, 512, 1.f, -1.f,
         nullptr, 0, S1, nullptr, 512, nullptr, nullptr, nullptr, nullptr, 0);
    // e_gen1  = x1 - s2@W1^T              -> bf16 EA2[:,0:512] ; S1 += e_gen1
    gemm(s2b, W1b, 512, 512, x1, 512, 1.f, -1.f,
         EA2, 768, S1, S1, 512, nullptr, nullptr, nullptr, nullptr, 0);
    // e_gen2  = x2 - y_target@W2^T        -> f32 S2
    gemm(ytb, W2b, 256, 512, x2, 512, 1.f, -1.f,
         nullptr, 0, S2, nullptr, 512, nullptr, nullptr, nullptr, nullptr, 0);
    // e_disc2 = x2 - s1@V1^T              -> bf16 EA1[:,1024:1536] ; S2 += e_disc2
    gemm(s1b, V1b, 512, 512, x2, 512, 1.f, -1.f,
         EA1 + 1024, 1536, S2, S2, 512, nullptr, nullptr, nullptr, nullptr, 0);
    // e_disc3 = y_target - s2@V2^T        -> bf16 EA2[:,512:768]
    gemm(s2b, V2b, 512, 256, y_target, 256, 1.f, -1.f,
         EA2 + 512, 768, nullptr, nullptr, 0, nullptr, nullptr, nullptr, nullptr, 0);

    // Phase 2 + LIF (fused): total = A@W^T - S ; LIF -> out
    // total1 = e_gen0@Eg0^T + e_disc2@Ed1^T - (e_gen1+e_disc1)
    gemm(EA1, WC1, 1536, 512, S1, 512, -1.f, 1.f,
         nullptr, 0, nullptr, nullptr, 0, j1, v1, x1, out, 0);
    // total2 = e_gen1@Eg1^T + e_disc3@Ed2^T - (e_gen2+e_disc2)
    gemm(EA2, WC2, 768, 512, S2, 512, -1.f, 1.f,
         nullptr, 0, nullptr, nullptr, 0, j2, v2, x2, out, 2048);
}
